// LSTM_Critic_21509196218919
// MI455X (gfx1250) — matrix operand order repaired and checked by reading, not compiled, not run on hardware
//
#include <hip/hip_runtime.h>
#include <hip/hip_bf16.h>

typedef __attribute__((ext_vector_type(16))) __bf16 v16bf;
typedef __attribute__((ext_vector_type(8)))  float  v8f;

#define T_STEPS 1024
#define BATCH   512
#define IN_DIM  64
#define HIDDEN  128
#define BT      32       // batch rows per workgroup
#define THREADS 512      // 16 waves (wave32)

// ---- workspace layout (dwords): pre-swizzled bf16 WMMA B-fragments ----
// fragment = 32 lanes x 8 dwords (16 bf16/lane) = 256 dwords = 1KB
#define WIH0_OFF 0         // 32 ntiles * 2 ktiles  = 16384 dwords (64KB)
#define WHH0_OFF 16384     // 32 * 4 = 32768 dwords (128KB)
#define WIH1_OFF 49152     // 32 * 4 = 32768 dwords (128KB)
#define WHH1_OFF 81920     // 32 * 4 = 32768 dwords (128KB)
#define WS_DWORDS 114688   // 448KB total

// ---- LDS layout (bytes) ----
#define L_WHH0 0           // 128KB recurrent weights, LDS-resident
#define L_WHH1 131072      // 128KB
#define L_H0   262144      // [2][32][128] bf16 = 16KB (double-buffered)
#define L_H1   278528      // 16KB
#define L_X    294912      // [32][64] bf16 = 4KB
#define LDS_BYTES 299008   // 292KB <= 320KB per workgroup

__device__ __forceinline__ unsigned f2bf1(float f) {
  unsigned u = __float_as_uint(f);
  return (u + 0x7fffu + ((u >> 16) & 1u)) >> 16;   // round-to-nearest-even
}
__device__ __forceinline__ unsigned packbf(float a, float b) {
  return f2bf1(a) | (f2bf1(b) << 16);
}
__device__ __forceinline__ float bf2f(unsigned hu) { return __uint_as_float(hu << 16); }
__device__ __forceinline__ float sigf(float x)      { return 1.0f / (1.0f + __expf(-x)); }
__device__ __forceinline__ float tanh_fast(float x) { return 2.0f * sigf(2.0f * x) - 1.0f; }

union FragU { v16bf v; float4 q[2]; unsigned u[8]; };

// A-matrix fragment (16x32 bf16, M x K) from a row-major bf16 LDS buffer.
// ISA 7.12.2: lanes 0-15 hold M=lane, VGPR0-3 => K=kb..kb+7, VGPR4-7 => K=kb+16..kb+23;
// lanes 16-31 same M rows, K offset +8 / +24.  => two ds_load_b128 per lane.
__device__ __forceinline__ v16bf load_a_lds(const char* base, int rowBytes, int mrow, int kb, int lane) {
  const int l16 = lane & 15;
  const int hi  = lane >> 4;
  const char* p = base + (size_t)(mrow + l16) * rowBytes + (size_t)(kb + (hi ? 8 : 0)) * 2;
  FragU f;
  f.q[0] = *(const float4*)p;
  f.q[1] = *(const float4*)(p + 32);
  return f.v;
}
// B-matrix fragment from pre-swizzled storage: lane-major, 32B contiguous per lane.
__device__ __forceinline__ v16bf load_b_glb(const unsigned* __restrict__ frag, int lane) {
  FragU f;
  const float4* p = (const float4*)frag + lane * 2;
  f.q[0] = p[0]; f.q[1] = p[1];
  return f.v;
}
__device__ __forceinline__ v16bf load_b_lds(const char* frag, int lane) {
  FragU f;
  const float4* p = (const float4*)frag + lane * 2;
  f.q[0] = p[0]; f.q[1] = p[1];
  return f.v;
}

// ---------------------------------------------------------------------------
// Prepass: convert f32 weights -> bf16, pre-swizzled into the exact WMMA
// B-fragment (32x16, KxN) per-lane layout so hot-loop loads are contiguous.
// B layout: lanes 0-15: N=lane, VGPR r => K = kt*32 + 2r (+1) (+8 for r>=4);
//           lanes 16-31: N=lane-16,    K = kt*32 + 8 + 2r (+1) (+8 for r>=4).
// ---------------------------------------------------------------------------
__global__ void swizzle_weights_kernel(const float* __restrict__ Wih0,
                                       const float* __restrict__ Whh0,
                                       const float* __restrict__ Wih1,
                                       const float* __restrict__ Whh1,
                                       unsigned* __restrict__ ws) {
  int id = blockIdx.x * 256 + threadIdx.x;
  if (id >= WS_DWORDS) return;
  const float* src; int Kt, Ksrc, base;
  if (id < WHH0_OFF)      { src = Wih0; Kt = 2; Ksrc = 64;  base = WIH0_OFF; }
  else if (id < WIH1_OFF) { src = Whh0; Kt = 4; Ksrc = 128; base = WHH0_OFF; }
  else if (id < WHH1_OFF) { src = Wih1; Kt = 4; Ksrc = 128; base = WIH1_OFF; }
  else                    { src = Whh1; Kt = 4; Ksrc = 128; base = WHH1_OFF; }
  int lid  = id - base;
  int r    = lid & 7;
  int lane = (lid >> 3) & 31;
  int kt   = (lid >> 8) % Kt;
  int nt   = lid / (256 * Kt);
  int n = nt * 16 + (lane & 15);
  int k = kt * 32 + ((lane >> 4) ? 8 : 0) + 2 * r + ((r >= 4) ? 8 : 0);
  ws[id] = packbf(src[n * Ksrc + k], src[n * Ksrc + k + 1]);
}

// ---------------------------------------------------------------------------
// Main persistent scan kernel: one WG per 32 batch rows, all 1024 timesteps.
// ---------------------------------------------------------------------------
__global__ __launch_bounds__(THREADS) void lstm_scan_kernel(
    const float* __restrict__ state, const unsigned* __restrict__ ws,
    const float* __restrict__ bih0, const float* __restrict__ bhh0,
    const float* __restrict__ bih1, const float* __restrict__ bhh1,
    const float* __restrict__ Wout, const float* __restrict__ boutp,
    float* __restrict__ out) {
  extern __shared__ char smem[];
  const int tid   = threadIdx.x;
  const int lane  = tid & 31;
  const int wave  = tid >> 5;
  const int l16   = lane & 15;
  const int mt    = wave >> 3;    // batch row tile (0..1)
  const int g     = wave & 7;     // hidden column tile (0..7)
  const int bbase = blockIdx.x * BT;

  char* whh0 = smem + L_WHH0;
  char* whh1 = smem + L_WHH1;
  char* h0b  = smem + L_H0;
  char* h1b  = smem + L_H1;
  char* xb   = smem + L_X;

  // Stage recurrent weight fragments into LDS (2 x 128KB), zero h state.
  {
    const float4* s0 = (const float4*)(ws + WHH0_OFF);
    const float4* s1 = (const float4*)(ws + WHH1_OFF);
    float4* d0 = (float4*)whh0;
    float4* d1 = (float4*)whh1;
#pragma unroll
    for (int i = 0; i < 16; ++i) {
      d0[tid + i * THREADS] = s0[tid + i * THREADS];
      d1[tid + i * THREADS] = s1[tid + i * THREADS];
    }
    float4 z = make_float4(0.f, 0.f, 0.f, 0.f);
    float4* hz = (float4*)h0b;                 // h0 + h1 contiguous: 2048 float4
#pragma unroll
    for (int i = 0; i < 4; ++i) hz[tid + i * THREADS] = z;
  }

  // Per-lane gate biases (gate order i,f,g,o along N).
  const int jcol = g * 16 + l16;               // hidden column owned by this lane
  float b0g[4], b1g[4];
#pragma unroll
  for (int gate = 0; gate < 4; ++gate) {
    int N = gate * HIDDEN + jcol;
    b0g[gate] = bih0[N] + bhh0[N];
    b1g[gate] = bih1[N] + bhh1[N];
  }
  // Value-head registers.
  const int vrow = tid >> 4, vchunk = tid & 15;
  float wreg[8];
#pragma unroll
  for (int i = 0; i < 8; ++i) wreg[i] = Wout[vchunk * 8 + i];
  const float bout = boutp[0];

  float c0[8], c1[8];
#pragma unroll
  for (int i = 0; i < 8; ++i) { c0[i] = 0.f; c1[i] = 0.f; }

  const int mrow  = mt * 16;
  const int mhalf = (lane >> 4) ? 8 : 0;       // C/D layout: VGPR v -> M = mhalf + v

  __syncthreads();

  for (int t = 0; t < T_STEPS; ++t) {
    // Fresh zero each iteration (opaque to the optimizer): folded into every
    // weight-fragment address so LICM cannot hoist the 24 loop-invariant
    // global fragments (192 VGPRs) out of the loop and spill them to scratch.
    unsigned zoff;
    asm volatile("s_mov_b32 %0, 0" : "=s"(zoff));

    const int cur = t & 1, nxt = cur ^ 1;
    char* h0c = h0b + cur * 8192;
    char* h0n = h0b + nxt * 8192;
    char* h1c = h1b + cur * 8192;
    char* h1n = h1b + nxt * 8192;

    // Stage x_t tile: 32x64 f32 -> bf16 LDS, exactly one float4 per thread.
    {
      const float4* xs = (const float4*)(state + ((size_t)t * BATCH + bbase) * IN_DIM);
      float4 xv = xs[tid];
      unsigned* xd = (unsigned*)xb + tid * 2;
      xd[0] = packbf(xv.x, xv.y);
      xd[1] = packbf(xv.z, xv.w);
      if (t + 1 < T_STEPS) __builtin_prefetch(&xs[tid + (BATCH * IN_DIM) / 4], 0, 1);
    }
    __syncthreads();

    // ---- layer 0: gates = x @ Wih0^T + h0 @ Whh0^T ----
    {
      v8f acc[4] = {};
#pragma unroll
      for (int kt = 0; kt < 2; ++kt) {                       // x part, K=0..63
        v16bf A = load_a_lds(xb, IN_DIM * 2, mrow, kt * 32, lane);
        v16bf B0 = load_b_glb(ws + zoff + (((0 * 8 + g) * 2 + kt) * 256), lane);
        v16bf B1 = load_b_glb(ws + zoff + (((1 * 8 + g) * 2 + kt) * 256), lane);
        v16bf B2 = load_b_glb(ws + zoff + (((2 * 8 + g) * 2 + kt) * 256), lane);
        v16bf B3 = load_b_glb(ws + zoff + (((3 * 8 + g) * 2 + kt) * 256), lane);
        acc[0] = __builtin_amdgcn_wmma_f32_16x16x32_bf16(false, A, false, B0, (short)0, acc[0], false, false);
        acc[1] = __builtin_amdgcn_wmma_f32_16x16x32_bf16(false, A, false, B1, (short)0, acc[1], false, false);
        acc[2] = __builtin_amdgcn_wmma_f32_16x16x32_bf16(false, A, false, B2, (short)0, acc[2], false, false);
        acc[3] = __builtin_amdgcn_wmma_f32_16x16x32_bf16(false, A, false, B3, (short)0, acc[3], false, false);
      }
#pragma unroll
      for (int kt = 0; kt < 4; ++kt) {                       // h part, K=0..127
        v16bf A = load_a_lds(h0c, HIDDEN * 2, mrow, kt * 32, lane);
        v16bf B0 = load_b_lds(whh0 + zoff + (size_t)(((0 * 8 + g) * 4 + kt) * 1024), lane);
        v16bf B1 = load_b_lds(whh0 + zoff + (size_t)(((1 * 8 + g) * 4 + kt) * 1024), lane);
        v16bf B2 = load_b_lds(whh0 + zoff + (size_t)(((2 * 8 + g) * 4 + kt) * 1024), lane);
        v16bf B3 = load_b_lds(whh0 + zoff + (size_t)(((3 * 8 + g) * 4 + kt) * 1024), lane);
        acc[0] = __builtin_amdgcn_wmma_f32_16x16x32_bf16(false, A, false, B0, (short)0, acc[0], false, false);
        acc[1] = __builtin_amdgcn_wmma_f32_16x16x32_bf16(false, A, false, B1, (short)0, acc[1], false, false);
        acc[2] = __builtin_amdgcn_wmma_f32_16x16x32_bf16(false, A, false, B2, (short)0, acc[2], false, false);
        acc[3] = __builtin_amdgcn_wmma_f32_16x16x32_bf16(false, A, false, B3, (short)0, acc[3], false, false);
      }
      unsigned short* hd = (unsigned short*)h0n;
#pragma unroll
      for (int v = 0; v < 8; ++v) {
        float iv = sigf(acc[0][v] + b0g[0]);
        float fv = sigf(acc[1][v] + b0g[1]);
        float gv = tanh_fast(acc[2][v] + b0g[2]);
        float ov = sigf(acc[3][v] + b0g[3]);
        float cn = fv * c0[v] + iv * gv;
        c0[v] = cn;
        hd[(mrow + mhalf + v) * HIDDEN + jcol] = (unsigned short)f2bf1(ov * tanh_fast(cn));
      }
    }
    __syncthreads();

    // ---- layer 1: gates = h0_new @ Wih1^T + h1 @ Whh1^T ----
    {
      v8f acc[4] = {};
#pragma unroll
      for (int kt = 0; kt < 4; ++kt) {                       // input = h0_new
        v16bf A = load_a_lds(h0n, HIDDEN * 2, mrow, kt * 32, lane);
        v16bf B0 = load_b_glb(ws + WIH1_OFF + zoff + (((0 * 8 + g) * 4 + kt) * 256), lane);
        v16bf B1 = load_b_glb(ws + WIH1_OFF + zoff + (((1 * 8 + g) * 4 + kt) * 256), lane);
        v16bf B2 = load_b_glb(ws + WIH1_OFF + zoff + (((2 * 8 + g) * 4 + kt) * 256), lane);
        v16bf B3 = load_b_glb(ws + WIH1_OFF + zoff + (((3 * 8 + g) * 4 + kt) * 256), lane);
        acc[0] = __builtin_amdgcn_wmma_f32_16x16x32_bf16(false, A, false, B0, (short)0, acc[0], false, false);
        acc[1] = __builtin_amdgcn_wmma_f32_16x16x32_bf16(false, A, false, B1, (short)0, acc[1], false, false);
        acc[2] = __builtin_amdgcn_wmma_f32_16x16x32_bf16(false, A, false, B2, (short)0, acc[2], false, false);
        acc[3] = __builtin_amdgcn_wmma_f32_16x16x32_bf16(false, A, false, B3, (short)0, acc[3], false, false);
      }
#pragma unroll
      for (int kt = 0; kt < 4; ++kt) {                       // recurrent h1
        v16bf A = load_a_lds(h1c, HIDDEN * 2, mrow, kt * 32, lane);
        v16bf B0 = load_b_lds(whh1 + zoff + (size_t)(((0 * 8 + g) * 4 + kt) * 1024), lane);
        v16bf B1 = load_b_lds(whh1 + zoff + (size_t)(((1 * 8 + g) * 4 + kt) * 1024), lane);
        v16bf B2 = load_b_lds(whh1 + zoff + (size_t)(((2 * 8 + g) * 4 + kt) * 1024), lane);
        v16bf B3 = load_b_lds(whh1 + zoff + (size_t)(((3 * 8 + g) * 4 + kt) * 1024), lane);
        acc[0] = __builtin_amdgcn_wmma_f32_16x16x32_bf16(false, A, false, B0, (short)0, acc[0], false, false);
        acc[1] = __builtin_amdgcn_wmma_f32_16x16x32_bf16(false, A, false, B1, (short)0, acc[1], false, false);
        acc[2] = __builtin_amdgcn_wmma_f32_16x16x32_bf16(false, A, false, B2, (short)0, acc[2], false, false);
        acc[3] = __builtin_amdgcn_wmma_f32_16x16x32_bf16(false, A, false, B3, (short)0, acc[3], false, false);
      }
      unsigned short* hd = (unsigned short*)h1n;
#pragma unroll
      for (int v = 0; v < 8; ++v) {
        float iv = sigf(acc[0][v] + b1g[0]);
        float fv = sigf(acc[1][v] + b1g[1]);
        float gv = tanh_fast(acc[2][v] + b1g[2]);
        float ov = sigf(acc[3][v] + b1g[3]);
        float cn = fv * c1[v] + iv * gv;
        c1[v] = cn;
        hd[(mrow + mhalf + v) * HIDDEN + jcol] = (unsigned short)f2bf1(ov * tanh_fast(cn));
      }
    }
    __syncthreads();

    // ---- value head: v[b] = h1_new[b,:] . Wout + bout ----
    {
      const unsigned* hp = (const unsigned*)h1n + vrow * (HIDDEN / 2) + vchunk * 4;
      float s = 0.f;
#pragma unroll
      for (int i = 0; i < 4; ++i) {
        unsigned u = hp[i];
        s += bf2f(u & 0xffffu) * wreg[2 * i] + bf2f(u >> 16) * wreg[2 * i + 1];
      }
      s += __shfl_xor(s, 1);
      s += __shfl_xor(s, 2);
      s += __shfl_xor(s, 4);
      s += __shfl_xor(s, 8);
      if (vchunk == 0) out[(size_t)t * BATCH + bbase + vrow] = s + bout;
    }
  }
}

extern "C" void kernel_launch(void* const* d_in, const int* in_sizes, int n_in,
                              void* d_out, int out_size, void* d_ws, size_t ws_size,
                              hipStream_t stream) {
  (void)in_sizes; (void)n_in; (void)out_size;
  if (ws_size < (size_t)WS_DWORDS * 4) return;
  const float* state = (const float*)d_in[0];
  const float* Wih0  = (const float*)d_in[1];
  const float* Whh0  = (const float*)d_in[2];
  const float* bih0  = (const float*)d_in[3];
  const float* bhh0  = (const float*)d_in[4];
  const float* Wih1  = (const float*)d_in[5];
  const float* Whh1  = (const float*)d_in[6];
  const float* bih1  = (const float*)d_in[7];
  const float* bhh1  = (const float*)d_in[8];
  const float* Wout  = (const float*)d_in[9];
  const float* bout  = (const float*)d_in[10];
  unsigned* ws = (unsigned*)d_ws;
  float* out = (float*)d_out;

  swizzle_weights_kernel<<<(WS_DWORDS + 255) / 256, 256, 0, stream>>>(
      Wih0, Whh0, Wih1, Whh1, ws);
  lstm_scan_kernel<<<BATCH / BT, THREADS, LDS_BYTES, stream>>>(
      state, ws, bih0, bhh0, bih1, bhh1, Wout, bout, out);
}
